// GFTLayer_65695819760080
// MI455X (gfx1250) — matrix operand order repaired and checked by reading, not compiled, not run on hardware
//
#include <hip/hip_runtime.h>

// Problem constants (match reference)
constexpr int NB = 64;    // batch of graphs
constexpr int NN = 512;   // nodes
constexpr int NC = 128;   // channels
constexpr int NE = 8192;  // edges
constexpr int NK = 128;   // k_nodes (top-k)

typedef _Float16 half_t;
typedef __attribute__((ext_vector_type(16))) _Float16 v16h;
typedef __attribute__((ext_vector_type(8)))  _Float16 v8h;
typedef __attribute__((ext_vector_type(4)))  _Float16 v4h;
typedef __attribute__((ext_vector_type(8)))  float    v8f;

// ---------------------------------------------------------------------------
// Batched GEMM:  C[b] = alpha * A[b](MxK) @ B[b](KxN)  (+ I if addI)
// f16 inputs, f32 WMMA accumulation, optional f16 and/or f32 outputs.
// 128x128 block tile, BK=32 (one v_wmma_f32_16x16x32_f16 K-step),
// 8 waves (wave32), each wave computes 64x32 via 4x2 16x16 accumulators.
// Double-buffered LDS; B staged transposed via packed 64-bit DS stores.
// ---------------------------------------------------------------------------
constexpr int BM = 128, BN = 128, BK = 32, LDT = 40; // LDT pad keeps 16B align

__global__ __launch_bounds__(256) void gemm_f16_wmma(
    const half_t* __restrict__ Ag, const half_t* __restrict__ Bg,
    half_t* __restrict__ Ch, float* __restrict__ Cf,
    int M, int Nd, int Kd, long sA, long sB, long sC,
    float alpha, int addI)
{
  __shared__ __align__(32) half_t Asl[2][BM][LDT];
  __shared__ __align__(32) half_t Bsl[2][BN][LDT]; // transposed: Bsl[buf][n][k]

  const int bz = blockIdx.z;
  const half_t* A  = Ag + (long)bz * sA;
  const half_t* Bm = Bg + (long)bz * sB;
  const int tileM = blockIdx.y * BM;
  const int tileN = blockIdx.x * BN;

  const int tid  = threadIdx.x;
  const int lane = tid & 31;
  const int wid  = tid >> 5;          // 0..7
  const int wRow = (wid & 1) * 64;    // 2 waves along M
  const int wCol = (wid >> 1) * 32;   // 4 waves along N
  const int lane16   = lane & 15;
  const int laneHalf = lane >> 4;

  v8f acc[4][2];
#pragma unroll
  for (int i = 0; i < 4; i++)
#pragma unroll
    for (int j = 0; j < 2; j++)
#pragma unroll
      for (int r = 0; r < 8; r++) acc[i][j][r] = 0.0f;

  // Stage assignments (256 threads)
  const int arow = tid >> 1;          // A: 0..127 rows
  const int acol = (tid & 1) * 16;    // A: 0 / 16
  const int bn4  = (tid & 31) * 4;    // B: n-group of 4 (coalesced along N)
  const int bk4  = (tid >> 5) * 4;    // B: k-group of 4

  // Per-thread staging registers for the in-flight tile
  v8h a0, a1;
  v4h br[4];

  auto loadTile = [&](int k0) {
    const half_t* aSrc = A + (long)(tileM + arow) * Kd + (k0 + acol);
    a0 = *(const v8h*)aSrc;
    a1 = *(const v8h*)(aSrc + 8);
    const half_t* bSrc = Bm + (long)(k0 + bk4) * Nd + (tileN + bn4);
#pragma unroll
    for (int r = 0; r < 4; r++)
      br[r] = *(const v4h*)(bSrc + (long)r * Nd);
  };
  auto storeTile = [&](int buf) {
    *(v8h*)&Asl[buf][arow][acol]     = a0;
    *(v8h*)&Asl[buf][arow][acol + 8] = a1;
    // transpose 4x4 micro-block: four packed 64-bit column stores
#pragma unroll
    for (int c = 0; c < 4; c++) {
      v4h col;
      col[0] = br[0][c]; col[1] = br[1][c]; col[2] = br[2][c]; col[3] = br[3][c];
      *(v4h*)&Bsl[buf][bn4 + c][bk4] = col;
    }
  };

  const int nIter = Kd / BK;
  loadTile(0);
  storeTile(0);

  for (int it = 0; it < nIter; ++it) {
    __syncthreads();
    const int cur = it & 1;
    if (it + 1 < nIter) loadTile((it + 1) * BK);
    if (it + 2 < nIter) {  // gfx1250 global_prefetch_b8, two tiles ahead
      __builtin_prefetch(A + (long)(tileM + arow) * Kd + ((it + 2) * BK + acol), 0, 1);
      __builtin_prefetch(Bm + (long)((it + 2) * BK + bk4) * Nd + (tileN + bn4), 0, 1);
    }

    // A fragments: 16x32 per tile; lane holds row=lane%16,
    // K chunks [h*8, h*8+7] and [16+h*8, 16+h*8+7] with h = lane/16
    v16h af[4];
#pragma unroll
    for (int i = 0; i < 4; i++) {
      const int row = wRow + i * 16 + lane16;
      v8h lo = *(const v8h*)&Asl[cur][row][laneHalf * 8];
      v8h hi = *(const v8h*)&Asl[cur][row][16 + laneHalf * 8];
      af[i] = __builtin_shufflevector(lo, hi, 0,1,2,3,4,5,6,7,8,9,10,11,12,13,14,15);
    }
    // B fragments: 32x16; lane holds col=lane%16, K chunks [h*8, h*8+7] and [16+h*8, 16+h*8+7]
    v16h bf[2];
#pragma unroll
    for (int j = 0; j < 2; j++) {
      const int col = wCol + j * 16 + lane16;
      v8h lo = *(const v8h*)&Bsl[cur][col][laneHalf * 8];
      v8h hi = *(const v8h*)&Bsl[cur][col][16 + laneHalf * 8];
      bf[j] = __builtin_shufflevector(lo, hi, 0,1,2,3,4,5,6,7,8,9,10,11,12,13,14,15);
    }

#pragma unroll
    for (int i = 0; i < 4; i++)
#pragma unroll
      for (int j = 0; j < 2; j++)
        acc[i][j] = __builtin_amdgcn_wmma_f32_16x16x32_f16(
            false, af[i], false, bf[j], (short)0, acc[i][j], false, false);

    if (it + 1 < nIter) storeTile(cur ^ 1);
  }

  // Epilogue: D layout — VGPR r: lanes0-15 -> M=r, lanes16-31 -> M=8+r; N=lane%16
#pragma unroll
  for (int i = 0; i < 4; i++)
#pragma unroll
    for (int j = 0; j < 2; j++) {
      const int col = tileN + wCol + j * 16 + lane16;
#pragma unroll
      for (int r = 0; r < 8; r++) {
        const int row = tileM + wRow + i * 16 + laneHalf * 8 + r;
        float v = alpha * acc[i][j][r];
        if (addI && row == col) v += 1.0f;
        const long off = (long)bz * sC + (long)row * Nd + col;
        if (Ch) Ch[off] = (half_t)v;
        if (Cf) Cf[off] = v;
      }
    }
}

// ---------------------------------------------------------------------------
// Elementwise / graph kernels
// ---------------------------------------------------------------------------
__global__ void scatter_edges(const int* __restrict__ ei, half_t* __restrict__ A) {
  int t = blockIdx.x * blockDim.x + threadIdx.x;
  if (t >= NB * NE) return;
  int b = t / NE, e = t % NE;
  int src = ei[(b * 2 + 0) * NE + e];
  int dst = ei[(b * 2 + 1) * NE + e];
  A[(long)b * NN * NN + (long)src * NN + dst] = (half_t)1.0f;
}

__global__ void degree_kernel(const half_t* __restrict__ A,
                              float* __restrict__ dis, float* __restrict__ sd) {
  int t = blockIdx.x * blockDim.x + threadIdx.x;
  if (t >= NB * NN) return;
  const half_t* row = A + (long)t * NN;
  float d = 0.f;
  for (int j = 0; j < NN; j++) d += (float)row[j];
  dis[t] = (d > 0.f) ? rsqrtf(d) : 0.f;
  sd[t]  = rsqrtf(d + 1.0f);      // self-loop degree = deg+1 > 0 always
}

// X = (-L)/4 = (D^-1/2 A D^-1/2 - I)/4 ;  R0 = I + X/8 (Horner innermost)
__global__ void build_XR(const half_t* __restrict__ A, const float* __restrict__ dis,
                         half_t* __restrict__ X, half_t* __restrict__ R) {
  int t = blockIdx.x * blockDim.x + threadIdx.x;
  if (t >= NB * NN * NN) return;
  int b = t / (NN * NN); int rem = t % (NN * NN);
  int i = rem / NN, j = rem % NN;
  float a = (float)A[t];
  float dlt = (i == j) ? 1.f : 0.f;
  float x = 0.25f * (dis[b * NN + i] * a * dis[b * NN + j] - dlt);
  X[t] = (half_t)x;
  R[t] = (half_t)(dlt + 0.125f * x);
}

// S = (D+I)^-1/2 (A + I) (D+I)^-1/2
__global__ void build_S(const half_t* __restrict__ A, const float* __restrict__ sd,
                        half_t* __restrict__ S) {
  int t = blockIdx.x * blockDim.x + threadIdx.x;
  if (t >= NB * NN * NN) return;
  int b = t / (NN * NN); int rem = t % (NN * NN);
  int i = rem / NN, j = rem % NN;
  float a = (float)A[t] + ((i == j) ? 1.f : 0.f);
  S[t] = (half_t)(sd[b * NN + i] * a * sd[b * NN + j]);
}

__global__ void f32_to_f16(const float* __restrict__ in, half_t* __restrict__ out, int n) {
  int t = blockIdx.x * blockDim.x + threadIdx.x;
  if (t < n) out[t] = (half_t)in[t];
}

__global__ void scores_kernel(const float* __restrict__ smooth,
                              const float* __restrict__ W, float* __restrict__ sc) {
  int t = blockIdx.x * blockDim.x + threadIdx.x;
  if (t >= NB * NN) return;
  const float* row = smooth + (long)t * NC;
  float acc = 0.f;
  for (int c = 0; c < NC; c++) acc += row[c] * W[c];
  sc[t] = 1.f / (1.f + __expf(-acc));
}

// rank-based top-K: rank_i = #{j : s_j > s_i} + #{j : s_j == s_i && j < i}
// exactly reproduces lax.top_k ordering (descending, low-index tie break).
__global__ void topk_kernel(const float* __restrict__ sc, int* __restrict__ idx) {
  int t = blockIdx.x * blockDim.x + threadIdx.x;
  if (t >= NB * NN) return;
  int b = t / NN, i = t % NN;
  const float* s = sc + b * NN;
  float si = s[i];
  int cnt = 0;
  for (int j = 0; j < NN; j++) {
    float sj = s[j];
    cnt += (sj > si) || (sj == si && j < i);
  }
  if (cnt < NK) idx[b * NK + cnt] = i;
}

__global__ void gather_x(const float* __restrict__ xenh, const int* __restrict__ idx,
                         float* __restrict__ out) {
  int t = blockIdx.x * blockDim.x + threadIdx.x;
  if (t >= NB * NK * NC) return;
  int b = t / (NK * NC);
  int r = (t / NC) % NK;
  int c = t % NC;
  out[t] = xenh[(long)b * NN * NC + (long)idx[b * NK + r] * NC + c];
}

__global__ void gather_adj(const half_t* __restrict__ A, const int* __restrict__ idx,
                           float* __restrict__ out) {
  int t = blockIdx.x * blockDim.x + threadIdx.x;
  if (t >= NB * NK * NK) return;
  int b = t / (NK * NK);
  int r = (t / NK) % NK;
  int c = t % NK;
  out[t] = (float)A[(long)b * NN * NN + (long)idx[b * NK + r] * NN + idx[b * NK + c]];
}

// ---------------------------------------------------------------------------
// Host side
// ---------------------------------------------------------------------------
static inline void launch_gemm(const half_t* A, const half_t* Bm, half_t* Ch, float* Cf,
                               int M, int Nd, int Kd, float alpha, int addI,
                               hipStream_t s) {
  dim3 grid(Nd / BN, M / BM, NB);
  gemm_f16_wmma<<<grid, 256, 0, s>>>(A, Bm, Ch, Cf, M, Nd, Kd,
                                     (long)M * Kd, (long)Kd * Nd, (long)M * Nd,
                                     alpha, addI);
}

extern "C" void kernel_launch(void* const* d_in, const int* in_sizes, int n_in,
                              void* d_out, int out_size, void* d_ws, size_t ws_size,
                              hipStream_t stream) {
  (void)in_sizes; (void)n_in; (void)out_size; (void)ws_size;
  const float* x  = (const float*)d_in[0];
  const int*   ei = (const int*)d_in[1];   // jax x64 off -> int32
  const float* W  = (const float*)d_in[2];
  float* out = (float*)d_out;

  char* ws = (char*)d_ws;
  auto alloc = [&](size_t bytes) -> char* {
    char* p = ws;
    ws += (bytes + 255) & ~(size_t)255;
    return p;
  };
  const size_t NN2 = (size_t)NB * NN * NN;
  half_t* Ah     = (half_t*)alloc(NN2 * 2);               // adjacency (kept to end)
  half_t* Xh     = (half_t*)alloc(NN2 * 2);               // X = -L/4 ; reused for S
  half_t* Ph     = (half_t*)alloc(NN2 * 2);               // Horner ping
  half_t* Qh     = (half_t*)alloc(NN2 * 2);               // Horner pong
  half_t* Xin    = (half_t*)alloc((size_t)NB * NN * NC * 2);
  half_t* xenh16 = (half_t*)alloc((size_t)NB * NN * NC * 2);
  float*  xenh32 = (float*) alloc((size_t)NB * NN * NC * 4);
  float*  smooth = (float*) alloc((size_t)NB * NN * NC * 4);
  float*  dis    = (float*) alloc((size_t)NB * NN * 4);
  float*  sd     = (float*) alloc((size_t)NB * NN * 4);
  float*  sc     = (float*) alloc((size_t)NB * NN * 4);
  int*    idx    = (int*)   alloc((size_t)NB * NK * 4);

  const int T = 256;
  // 1) adjacency
  hipMemsetAsync(Ah, 0, NN2 * 2, stream);
  scatter_edges<<<(NB * NE + T - 1) / T, T, 0, stream>>>(ei, Ah);
  // 2) degrees -> D^-1/2 and (D+I)^-1/2
  degree_kernel<<<(NB * NN + T - 1) / T, T, 0, stream>>>(Ah, dis, sd);
  // 3) X = -L/4 and R0 = I + X/8
  build_XR<<<(NB * NN * NN + T - 1) / T, T, 0, stream>>>(Ah, dis, Xh, Ph);
  // 4) Horner: R <- I + (X @ R)/k  for k = 7..1   (exp(X) Taylor deg-8)
  for (int k = 7; k >= 1; --k) {
    launch_gemm(Xh, Ph, Qh, nullptr, NN, NN, NN, 1.0f / (float)k, 1, stream);
    half_t* tmp = Ph; Ph = Qh; Qh = tmp;
  }
  // 5) two squarings: E = exp(-L) = (exp(X))^4
  launch_gemm(Ph, Ph, Qh, nullptr, NN, NN, NN, 1.0f, 0, stream);
  launch_gemm(Qh, Qh, Ph, nullptr, NN, NN, NN, 1.0f, 0, stream);   // E in Ph
  // 6) x_enh = E @ x
  f32_to_f16<<<(NB * NN * NC + T - 1) / T, T, 0, stream>>>(x, Xin, NB * NN * NC);
  launch_gemm(Ph, Xin, xenh16, xenh32, NN, NC, NN, 1.0f, 0, stream);
  // 7) smooth = S @ x_enh   (S built into Xh, which is free now)
  build_S<<<(NB * NN * NN + T - 1) / T, T, 0, stream>>>(Ah, sd, Xh);
  launch_gemm(Xh, xenh16, nullptr, smooth, NN, NC, NN, 1.0f, 0, stream);
  // 8) scores = sigmoid(smooth @ W) ; top-K ; gathers
  scores_kernel<<<(NB * NN + T - 1) / T, T, 0, stream>>>(smooth, W, sc);
  topk_kernel<<<(NB * NN + T - 1) / T, T, 0, stream>>>(sc, idx);
  gather_x  <<<(NB * NK * NC + T - 1) / T, T, 0, stream>>>(xenh32, idx, out);
  gather_adj<<<(NB * NK * NK + T - 1) / T, T, 0, stream>>>(Ah, idx,
                                                           out + (size_t)NB * NK * NC);
}
